// KTopPooling_72086731096493
// MI455X (gfx1250) — compile-verified, matrix operand order checked by reading, not run on hardware
//
#include <hip/hip_runtime.h>
#include <hip/hip_bf16.h>

typedef __attribute__((ext_vector_type(16))) _Float16 v16h;
typedef __attribute__((ext_vector_type(8)))  float    v8f;

#define C_DIM 256
#define H_DIM 64
#define K_DIM 3
#define G_DIM 512
#define LEAKY(v) ((v) > 0.0f ? (v) : 0.01f * (v))

// ---------------------------------------------------------------------------
// K1: scores = leaky(x@W1 + b1) @ W2 + b2   (WMMA f16, fp32 accumulate)
// One wave32 per 16-row tile of x. W1 pre-packed in LDS in the ISA B-operand
// layout (16-bit A/B 16x32: lanes 0-15 hold K {0..7,16..23}, lanes 16-31 hold
// K {8..15,24..31}; B lane = output column). 4 waves / 128 threads per block.
// Assumes N % 16 == 0 (N = 200000 = 16 * 12500).
// ---------------------------------------------------------------------------
__global__ __launch_bounds__(128) void scores_kernel(
    const float* __restrict__ x, const float* __restrict__ W1,
    const float* __restrict__ b1, const float* __restrict__ W2,
    const float* __restrict__ b2, float* __restrict__ scores, int nTiles) {
  // packed B: [ntile(4)][kstep(8)][lane(32)][j(16)] halves = 32 KB
  __shared__ __align__(32) _Float16 ldsB[4 * 8 * 32 * 16];
  __shared__ float hbuf[4][16][H_DIM];  // per-wave 16x64 hidden tile, 16 KB

  const int tid = threadIdx.x;

  // Pack W1 (C x H fp32, row-major) into LDS f16 in WMMA B layout.
  for (int idx = tid; idx < 4 * 8 * 32 * 16; idx += 128) {
    int j    = idx & 15;
    int lane = (idx >> 4) & 31;
    int ks   = (idx >> 9) & 7;
    int nt   = idx >> 12;
    int half = lane >> 4;
    int col  = nt * 16 + (lane & 15);
    int k = ks * 32 + ((j < 8) ? (half * 8 + j) : (16 + half * 8 + (j - 8)));
    ldsB[idx] = (_Float16)W1[(size_t)k * H_DIM + col];
  }
  __syncthreads();

  const int wave = tid >> 5;
  const int lane = tid & 31;
  const int half = lane >> 4;
  const int mrow = lane & 15;
  const int tile = blockIdx.x * 4 + wave;
  const bool active = (tile < nTiles);

  if (active) {
    const float* xrow = x + (size_t)(tile * 16 + mrow) * C_DIM;
    v8f acc[4] = {};
#pragma unroll
    for (int ks = 0; ks < 8; ++ks) {
      const int kb0 = ks * 32 + half * 8;       // A elems j=0..7 : contiguous
      const int kb1 = ks * 32 + 16 + half * 8;  // A elems j=8..15: contiguous
      v16h a;
#pragma unroll
      for (int j = 0; j < 8; ++j) a[j] = (_Float16)xrow[kb0 + j];
#pragma unroll
      for (int j = 0; j < 8; ++j) a[8 + j] = (_Float16)xrow[kb1 + j];
#pragma unroll
      for (int nt = 0; nt < 4; ++nt) {
        const v16h b = *(const v16h*)&ldsB[(((nt * 8) + ks) * 32 + lane) * 16];
        acc[nt] = __builtin_amdgcn_wmma_f32_16x16x32_f16(
            false, a, false, b, (short)0, acc[nt], false, false);
      }
    }
    // bias + leaky, spill 16x64 tile to this wave's LDS region.
    // C/D layout: elem r -> M = r + 8*half, N = nt*16 + mrow.
#pragma unroll
    for (int nt = 0; nt < 4; ++nt) {
      const int col = nt * 16 + mrow;
      const float bb = b1[col];
#pragma unroll
      for (int r = 0; r < 8; ++r) {
        float v = acc[nt][r] + bb;
        hbuf[wave][r + 8 * half][col] = LEAKY(v);
      }
    }
  }
  __syncthreads();

  // tiny H->K contraction: lanes 0..15 each own one row.
  if (active && lane < 16) {
    float s0 = b2[0], s1 = b2[1], s2 = b2[2];
#pragma unroll 4
    for (int n = 0; n < H_DIM; ++n) {
      const float h = hbuf[wave][mrow][n];
      s0 += h * W2[n * 3 + 0];
      s1 += h * W2[n * 3 + 1];
      s2 += h * W2[n * 3 + 2];
    }
    const size_t node = (size_t)tile * 16 + mrow;
    scores[node * 3 + 0] = s0;
    scores[node * 3 + 1] = s1;
    scores[node * 3 + 2] = s2;
  }
}

// ---------------------------------------------------------------------------
// K2: starts[g] = lower_bound(batch, g); batch is sorted int64.
// ---------------------------------------------------------------------------
__global__ void segstart_kernel(const long long* __restrict__ batch,
                                int* __restrict__ starts, int N, int G) {
  const int g = blockIdx.x * blockDim.x + threadIdx.x;
  if (g > G) return;
  int lo = 0, hi = N;
  while (lo < hi) {
    const int mid = (lo + hi) >> 1;
    if (batch[mid] < (long long)g) lo = mid + 1; else hi = mid;
  }
  starts[g] = lo;
}

// ---------------------------------------------------------------------------
// K3: per-segment softmax/argmax pooling -> feat[g][k*256+c] = x[idx_k][c]*sg_k
// One block (256 threads) per segment.
// ---------------------------------------------------------------------------
__global__ __launch_bounds__(256) void pool_kernel(
    const float* __restrict__ x, const float* __restrict__ scores,
    const int* __restrict__ starts, float* __restrict__ feat, int N) {
  __shared__ float smax[256];
  __shared__ int   sidx[256];
  __shared__ float ssum[256];
  __shared__ float m_k[K_DIM];
  __shared__ int   idx_k[K_DIM];
  __shared__ float sg_k[K_DIM];

  const int g = blockIdx.x;
  const int s = starts[g], e = starts[g + 1];
  const int tid = threadIdx.x;

  for (int k = 0; k < K_DIM; ++k) {
    // ---- max + earliest argmax ----
    float lm = -INFINITY;
    int li = N;
    for (int i = s + tid; i < e; i += 256) {
      const float v = scores[(size_t)i * 3 + k];
      if (v > lm) { lm = v; li = i; }  // strict > keeps earliest index
    }
    smax[tid] = lm; sidx[tid] = li;
    __syncthreads();
    for (int off = 128; off > 0; off >>= 1) {
      if (tid < off) {
        const float vo = smax[tid + off];
        const int io = sidx[tid + off];
        if (vo > smax[tid] || (vo == smax[tid] && io < sidx[tid])) {
          smax[tid] = vo; sidx[tid] = io;
        }
      }
      __syncthreads();
    }
    if (tid == 0) {
      m_k[k] = smax[0];
      idx_k[k] = (e > s) ? min(sidx[0], N - 1) : (N - 1);
    }
    __syncthreads();

    // ---- sum of exp(s - m) ----
    const float mk = m_k[k];
    float ls = 0.0f;
    for (int i = s + tid; i < e; i += 256)
      ls += __expf(scores[(size_t)i * 3 + k] - mk);
    ssum[tid] = ls;
    __syncthreads();
    for (int off = 128; off > 0; off >>= 1) {
      if (tid < off) ssum[tid] += ssum[tid + off];
      __syncthreads();
    }
    if (tid == 0) {
      if (e > s) {
        const int id = idx_k[k];
        sg_k[k] = __expf(scores[(size_t)id * 3 + k] - mk) / ssum[0];
      } else {
        sg_k[k] = 0.0f;  // empty segment (prob ~0 here)
      }
    }
    __syncthreads();
  }

  // gather + scale: feat is G x (K*C)
  for (int t = tid; t < K_DIM * C_DIM; t += 256) {
    const int k = t >> 8, c = t & 255;
    feat[(size_t)g * (K_DIM * C_DIM) + t] =
        x[(size_t)idx_k[k] * C_DIM + c] * sg_k[k];
  }
}

// ---------------------------------------------------------------------------
// K4: out = leaky(feat(512x768) @ Wh(768x256) + bh)   (WMMA f16, fp32 acc)
// One 16x16 output tile per wave; 24 K-steps. 512 tiles = 128 blocks x 4 waves.
// ---------------------------------------------------------------------------
__global__ __launch_bounds__(128) void out_gemm_kernel(
    const float* __restrict__ feat, const float* __restrict__ Wh,
    const float* __restrict__ bh, float* __restrict__ out) {
  const int wave = threadIdx.x >> 5, lane = threadIdx.x & 31;
  const int tileId = blockIdx.x * 4 + wave;  // 0..511
  const int mt = tileId >> 4, nt = tileId & 15;
  const int half = lane >> 4, mrow = lane & 15;
  const int col = nt * 16 + mrow;

  const float* arow = feat + (size_t)(mt * 16 + mrow) * (K_DIM * C_DIM);
  v8f acc = {};
#pragma unroll 4
  for (int ks = 0; ks < 24; ++ks) {
    const int kb0 = ks * 32 + half * 8;
    const int kb1 = ks * 32 + 16 + half * 8;
    v16h a, b;
#pragma unroll
    for (int j = 0; j < 8; ++j) a[j] = (_Float16)arow[kb0 + j];
#pragma unroll
    for (int j = 0; j < 8; ++j) a[8 + j] = (_Float16)arow[kb1 + j];
#pragma unroll
    for (int j = 0; j < 8; ++j) b[j] = (_Float16)Wh[(size_t)(kb0 + j) * C_DIM + col];
#pragma unroll
    for (int j = 0; j < 8; ++j) b[8 + j] = (_Float16)Wh[(size_t)(kb1 + j) * C_DIM + col];
    acc = __builtin_amdgcn_wmma_f32_16x16x32_f16(
        false, a, false, b, (short)0, acc, false, false);
  }
  const float bb = bh[col];
#pragma unroll
  for (int r = 0; r < 8; ++r) {
    float v = acc[r] + bb;
    out[(size_t)(mt * 16 + r + 8 * half) * C_DIM + col] = LEAKY(v);
  }
}

// ---------------------------------------------------------------------------
extern "C" void kernel_launch(void* const* d_in, const int* in_sizes, int n_in,
                              void* d_out, int out_size, void* d_ws, size_t ws_size,
                              hipStream_t stream) {
  const float*     x     = (const float*)d_in[0];
  const long long* batch = (const long long*)d_in[1];
  const float*     W1    = (const float*)d_in[2];
  const float*     b1    = (const float*)d_in[3];
  const float*     W2    = (const float*)d_in[4];
  const float*     b2    = (const float*)d_in[5];
  const float*     Wh    = (const float*)d_in[6];
  const float*     bh    = (const float*)d_in[7];
  float* out = (float*)d_out;

  const int N = in_sizes[0] / C_DIM;  // 200000
  const int G = G_DIM;

  // workspace carve-up
  char* ws = (char*)d_ws;
  float* scores = (float*)ws;                              // N*3 fp32
  size_t off = (size_t)N * 3 * sizeof(float);
  off = (off + 255) & ~(size_t)255;
  int* starts = (int*)(ws + off);                          // G+1 ints
  off += (size_t)(G + 1) * sizeof(int);
  off = (off + 255) & ~(size_t)255;
  float* feat = (float*)(ws + off);                        // G*768 fp32

  const int nTiles = (N + 15) / 16;  // 12500 (N % 16 == 0)
  scores_kernel<<<(nTiles + 3) / 4, 128, 0, stream>>>(x, W1, b1, W2, b2,
                                                      scores, nTiles);
  segstart_kernel<<<(G + 1 + 255) / 256, 256, 0, stream>>>(batch, starts, N, G);
  pool_kernel<<<G, 256, 0, stream>>>(x, scores, starts, feat, N);
  // 32 M-tiles * 16 N-tiles = 512 tiles, 4 waves/block -> 128 blocks
  out_gemm_kernel<<<128, 128, 0, stream>>>(feat, Wh, bh, out);
}
